// ModalityAgnosticTransformer_87565793230869
// MI455X (gfx1250) — compile-verified, matrix operand order checked by reading, not run on hardware
//
#include <hip/hip_runtime.h>
#include <hip/hip_bf16.h>

// ---------------------------------------------------------------------------
// ViT-B/16 forward (B=32) for gfx1250 (MI455X).
// All GEMMs run through one bf16-WMMA kernel (v_wmma_f32_16x16x32_bf16),
// converting fp32 operands -> bf16 while staging into LDS.
// Token dim padded 197 -> NT=224 so every K is a multiple of 32.
// Requires ~305 MB of workspace (d_ws).
// ---------------------------------------------------------------------------

#define BB   32
#define DD   768
#define HH   12
#define HDIM 64
#define FFD  3072
#define NTOK 197
#define NT   224          // padded token count (multiple of 32)
#define ROWS (BB * NT)    // 7168
#define NPATCH 196
#define NCLS 1000

typedef __bf16 bf16;
typedef __bf16 bf16x8  __attribute__((ext_vector_type(8)));
typedef __bf16 bf16x16 __attribute__((ext_vector_type(16)));
typedef float  f32x4   __attribute__((ext_vector_type(4)));
typedef float  f32x8   __attribute__((ext_vector_type(8)));

#define LDS_K 40   // padded LDS row stride in halfs (32 data + 8 pad, 16B aligned)

// ---- stage a 128x32 fp32 tile (row-major, arbitrary ld) into LDS as bf16 ----
__device__ __forceinline__ void stage_tile(const float* __restrict__ g, int rows,
                                           int ld, int rowBase, int kb,
                                           bf16* s) {
  const int t  = threadIdx.x;      // 0..255
  const int r  = t >> 1;           // 0..127
  const int c0 = (t & 1) << 4;     // 0 or 16
  const int gr = rowBase + r;
  float v[16];
  if (gr < rows) {
    const float* p = g + (size_t)gr * ld + kb + c0;
#pragma unroll
    for (int i = 0; i < 16; i += 4) {
      f32x4 q = *(const f32x4*)(p + i);
      v[i + 0] = q.x; v[i + 1] = q.y; v[i + 2] = q.z; v[i + 3] = q.w;
    }
  } else {
#pragma unroll
    for (int i = 0; i < 16; i++) v[i] = 0.f;
  }
  bf16x8 h0, h1;
#pragma unroll
  for (int i = 0; i < 8; i++) { h0[i] = (bf16)v[i]; h1[i] = (bf16)v[i + 8]; }
  bf16* d = s + r * LDS_K + c0;
  *(bf16x8*)d       = h0;
  *(bf16x8*)(d + 8) = h1;
}

// ---- load one 16x32 bf16 WMMA operand fragment from LDS ----
// ISA layout: lane<16 -> row=lane,    halfs = K[0..7]  ++ K[16..23]
//             lane>=16 -> row=lane-16, halfs = K[8..15] ++ K[24..31]
__device__ __forceinline__ bf16x16 load_frag(const bf16* s, int rowBase, int lane) {
  const bf16* p = s + (rowBase + (lane & 15)) * LDS_K + ((lane >> 4) << 3);
  bf16x8 lo = *(const bf16x8*)p;
  bf16x8 hi = *(const bf16x8*)(p + 16);
  return __builtin_shufflevector(lo, hi, 0, 1, 2, 3, 4, 5, 6, 7,
                                 8, 9, 10, 11, 12, 13, 14, 15);
}

// ---------------------------------------------------------------------------
// Batched GEMM:  C[m,n] (+= res) = sum_k A[m,k] * W[n,k]  (+ bias[n], gelu)
// A: fp32 [M,K] row stride lda ; W: fp32 [N,K] row stride ldw (i.e. C = A @ W^T)
// Batch z -> (zo = z / batchInner, zi = z % batchInner); element-offsets via
// (sAo,sAi), (sWo,sWi), (sCo,sCi). flags bit0 = exact GELU.
// Block: 256 thr = 8 waves, tile 128x128, K-step 32; wave -> 64x32 (8 wmma).
// ---------------------------------------------------------------------------
__global__ __launch_bounds__(256) void gemm_bf16_wmma(
    const float* __restrict__ A, const float* __restrict__ W,
    const float* __restrict__ bias, const float* __restrict__ res,
    float* __restrict__ C,
    int M, int N, int K, int lda, int ldw, int ldc,
    int batchInner, long sAo, long sAi, long sWo, long sWi, long sCo, long sCi,
    int flags) {
  __shared__ bf16 sA[128 * LDS_K];
  __shared__ bf16 sW[128 * LDS_K];

  const int z  = blockIdx.z;
  const int zo = z / batchInner;
  const int zi = z - zo * batchInner;
  A += (size_t)zo * sAo + (size_t)zi * sAi;
  W += (size_t)zo * sWo + (size_t)zi * sWi;
  const size_t cOff = (size_t)zo * sCo + (size_t)zi * sCi;

  const int mBase = blockIdx.y * 128;
  const int nBase = blockIdx.x * 128;
  const int t    = threadIdx.x;
  const int lane = t & 31;
  const int wid  = t >> 5;
  const int wm   = (wid & 1) * 64;   // wave M offset inside block tile
  const int wn   = (wid >> 1) * 32;  // wave N offset inside block tile

  f32x8 acc[4][2];
#pragma unroll
  for (int i = 0; i < 4; i++)
#pragma unroll
    for (int j = 0; j < 2; j++)
#pragma unroll
      for (int v = 0; v < 8; v++) acc[i][j][v] = 0.f;

  for (int kb = 0; kb < K; kb += 32) {
    __syncthreads();
    stage_tile(A, M, lda, mBase, kb, sA);
    stage_tile(W, N, ldw, nBase, kb, sW);
    if (kb + 32 < K) {  // prefetch next K tile (global_prefetch_b8)
      const int r = t >> 1, c0 = (t & 1) << 4;
      if (mBase + r < M)
        __builtin_prefetch(A + (size_t)(mBase + r) * lda + kb + 32 + c0, 0, 1);
      if (nBase + r < N)
        __builtin_prefetch(W + (size_t)(nBase + r) * ldw + kb + 32 + c0, 0, 1);
    }
    __syncthreads();

    bf16x16 af[4], wf[2];
#pragma unroll
    for (int i = 0; i < 4; i++) af[i] = load_frag(sA, wm + i * 16, lane);
#pragma unroll
    for (int i = 0; i < 2; i++) wf[i] = load_frag(sW, wn + i * 16, lane);

#pragma unroll
    for (int mt = 0; mt < 4; mt++)
#pragma unroll
      for (int nt = 0; nt < 2; nt++)
        acc[mt][nt] = __builtin_amdgcn_wmma_f32_16x16x32_bf16(
            false, af[mt], false, wf[nt], (short)0, acc[mt][nt], false, false);
  }

  // Epilogue. C/D layout: VGPR v, lane L -> m = v + (L>=16 ? 8 : 0), n = L&15
  const int rowSel = (lane >> 4) * 8;
  const int colSel = lane & 15;
#pragma unroll
  for (int mt = 0; mt < 4; mt++) {
#pragma unroll
    for (int nt = 0; nt < 2; nt++) {
      const int n = nBase + wn + nt * 16 + colSel;
      if (n >= N) continue;
      const float bv = bias ? bias[n] : 0.f;
#pragma unroll
      for (int v = 0; v < 8; v++) {
        const int m = mBase + wm + mt * 16 + rowSel + v;
        if (m >= M) continue;
        float cv = acc[mt][nt][v] + bv;
        if (flags & 1)  // exact GELU: 0.5*x*(1+erf(x/sqrt(2)))
          cv = 0.5f * cv * (1.f + erff(cv * 0.70710678118654752f));
        const size_t idx = cOff + (size_t)m * ldc + n;
        if (res) cv += res[idx];
        C[idx] = cv;
      }
    }
  }
}

// ---- im2col: x[B,3,224,224] -> Ap[6272, 768] (k = c*256 + p*16 + q) ----
__global__ void im2col_kernel(const float* __restrict__ x, float* __restrict__ Ap) {
  size_t i = (size_t)blockIdx.x * 256 + threadIdx.x;
  if (i >= (size_t)BB * NPATCH * DD) return;
  int k = (int)(i % DD);
  int row = (int)(i / DD);
  int b = row / NPATCH, tt = row % NPATCH, hp = tt / 14, wp = tt % 14;
  int c = k >> 8, p = (k >> 4) & 15, q = k & 15;
  Ap[i] = x[((size_t)(b * 3 + c) * 224 + hp * 16 + p) * 224 + wp * 16 + q];
}

// ---- assemble tok[B,NT,768] = [cls | patches] + pos ; zero padding rows ----
__global__ void assemble_tok(const float* __restrict__ pout,
                             const float* __restrict__ cls,
                             const float* __restrict__ pos,
                             float* __restrict__ tok) {
  size_t i = (size_t)blockIdx.x * 256 + threadIdx.x;
  if (i >= (size_t)BB * NT * DD) return;
  int d = (int)(i % DD);
  int n = (int)((i / DD) % NT);
  int b = (int)(i / ((size_t)DD * NT));
  float v;
  if (n == 0)            v = cls[d] + pos[d];
  else if (n <= NPATCH)  v = pout[((size_t)b * NPATCH + (n - 1)) * DD + d] +
                             pos[(size_t)n * DD + d];
  else                   v = 0.f;
  tok[i] = v;
}

// ---- layernorm over 768, one block (256 thr) per row ----
__global__ __launch_bounds__(256) void layernorm_kernel(
    const float* __restrict__ x, const float* __restrict__ g,
    const float* __restrict__ b, float* __restrict__ y) {
  const int t = threadIdx.x;
  const size_t row = blockIdx.x;
  const float* xr = x + row * DD;
  float x0 = xr[t], x1 = xr[t + 256], x2 = xr[t + 512];
  __shared__ float sh[256];
  sh[t] = x0 + x1 + x2;
  __syncthreads();
  for (int o = 128; o > 0; o >>= 1) { if (t < o) sh[t] += sh[t + o]; __syncthreads(); }
  const float mean = sh[0] * (1.f / 768.f);
  __syncthreads();
  float d0 = x0 - mean, d1 = x1 - mean, d2 = x2 - mean;
  sh[t] = d0 * d0 + d1 * d1 + d2 * d2;
  __syncthreads();
  for (int o = 128; o > 0; o >>= 1) { if (t < o) sh[t] += sh[t + o]; __syncthreads(); }
  const float inv = rsqrtf(sh[0] * (1.f / 768.f) + 1e-5f);
  float* yr = y + row * DD;
  yr[t]       = d0 * inv * g[t]       + b[t];
  yr[t + 256] = d1 * inv * g[t + 256] + b[t + 256];
  yr[t + 512] = d2 * inv * g[t + 512] + b[t + 512];
}

// ---- softmax over key dim (in place), scale 1/8, mask keys >= 197 ----
__global__ __launch_bounds__(256) void softmax_kernel(float* __restrict__ S) {
  const int t = threadIdx.x;
  float* p = S + (size_t)blockIdx.x * NT;
  float v = (t < NTOK) ? p[t] * 0.125f : -3.4e38f;
  __shared__ float sh[256];
  sh[t] = v;
  __syncthreads();
  for (int o = 128; o > 0; o >>= 1) { if (t < o) sh[t] = fmaxf(sh[t], sh[t + o]); __syncthreads(); }
  const float mx = sh[0];
  __syncthreads();
  const float e = (t < NTOK) ? expf(v - mx) : 0.f;
  sh[t] = e;
  __syncthreads();
  for (int o = 128; o > 0; o >>= 1) { if (t < o) sh[t] += sh[t + o]; __syncthreads(); }
  const float inv = 1.f / sh[0];
  if (t < NT) p[t] = e * inv;
}

// ---- Vt[(b*12+h)*64 + d][j] = qkv[b*NT+j][1536 + h*64 + d] ----
__global__ void v_transpose(const float* __restrict__ qkv, float* __restrict__ Vt) {
  size_t i = (size_t)blockIdx.x * 256 + threadIdx.x;
  if (i >= (size_t)BB * HH * HDIM * NT) return;
  int j = (int)(i % NT);
  int d = (int)((i / NT) % HDIM);
  int z = (int)(i / ((size_t)NT * HDIM));
  int b = z / HH, h = z % HH;
  Vt[i] = qkv[((size_t)b * NT + j) * (3 * DD) + 2 * DD + h * HDIM + d];
}

// ---------------------------------------------------------------------------
static inline void launch_gemm(hipStream_t st, const float* A, const float* W,
                               const float* bias, const float* res, float* C,
                               int M, int N, int K, int lda, int ldw, int ldc,
                               int Z, int bi, long sAo, long sAi, long sWo,
                               long sWi, long sCo, long sCi, int flags) {
  dim3 g((N + 127) / 128, (M + 127) / 128, Z);
  gemm_bf16_wmma<<<g, dim3(256), 0, st>>>(A, W, bias, res, C, M, N, K, lda, ldw,
                                          ldc, bi, sAo, sAi, sWo, sWi, sCo, sCi,
                                          flags);
}

extern "C" void kernel_launch(void* const* d_in, const int* in_sizes, int n_in,
                              void* d_out, int out_size, void* d_ws,
                              size_t ws_size, hipStream_t stream) {
  (void)in_sizes; (void)n_in; (void)out_size; (void)ws_size;
  const float* x       = (const float*)d_in[0];
  const float* patch_w = (const float*)d_in[1];
  const float* patch_b = (const float*)d_in[2];
  const float* cls     = (const float*)d_in[3];
  const float* pos     = (const float*)d_in[4];
  const float* ln1_s   = (const float*)d_in[5];
  const float* ln1_b   = (const float*)d_in[6];
  const float* qkv_w   = (const float*)d_in[7];
  const float* qkv_b   = (const float*)d_in[8];
  const float* proj_w  = (const float*)d_in[9];
  const float* proj_b  = (const float*)d_in[10];
  const float* ln2_s   = (const float*)d_in[11];
  const float* ln2_b   = (const float*)d_in[12];
  const float* fc1_w   = (const float*)d_in[13];
  const float* fc1_b   = (const float*)d_in[14];
  const float* fc2_w   = (const float*)d_in[15];
  const float* fc2_b   = (const float*)d_in[16];
  const float* head_w  = (const float*)d_in[17];
  const float* head_b  = (const float*)d_in[18];
  float* out = (float*)d_out;

  // workspace layout (floats); total ~79.8M floats (~305 MB)
  float* w = (float*)d_ws;
  size_t o = 0;
  float* tok  = w + o; o += (size_t)ROWS * DD;          // 5,505,024
  float* lnb  = w + o; o += (size_t)ROWS * DD;          // 5,505,024
  float* qkvb = w + o; o += (size_t)ROWS * 3 * DD;      // 16,515,072
  float* Vt   = w + o; o += (size_t)BB * HH * HDIM * NT;// 5,505,024
  float* Sb   = w + o; o += (size_t)BB * HH * NT * NT;  // 19,267,584
  float* Ob   = w + o; o += (size_t)ROWS * DD;          // 5,505,024
  float* hb   = w + o; o += (size_t)ROWS * FFD;         // 22,020,096
  float* Ap   = hb;                                     // aliased (pre-layer)
  float* pout = hb + (size_t)BB * NPATCH * DD;          // aliased (pre-layer)

  const long ZA = 0;
  const int elemBlocks_tok = (int)(((size_t)BB * NT * DD + 255) / 256);
  const int elemBlocks_im  = (int)(((size_t)BB * NPATCH * DD + 255) / 256);
  const int elemBlocks_vt  = (int)(((size_t)BB * HH * HDIM * NT + 255) / 256);

  // 1) patch embedding: Ap = im2col(x); pout = Ap @ patch_w^T + patch_b
  im2col_kernel<<<elemBlocks_im, 256, 0, stream>>>(x, Ap);
  launch_gemm(stream, Ap, patch_w, patch_b, nullptr, pout,
              BB * NPATCH, DD, DD, DD, DD, DD, 1, 1, ZA, ZA, ZA, ZA, ZA, ZA, 0);
  assemble_tok<<<elemBlocks_tok, 256, 0, stream>>>(pout, cls, pos, tok);

  for (int d = 0; d < 12; d++) {
    // ---- attention ----
    layernorm_kernel<<<ROWS, 256, 0, stream>>>(tok, ln1_s + d * DD,
                                               ln1_b + d * DD, lnb);
    launch_gemm(stream, lnb, qkv_w + (size_t)d * 3 * DD * DD,
                qkv_b + (size_t)d * 3 * DD, nullptr, qkvb,
                ROWS, 3 * DD, DD, DD, DD, 3 * DD, 1, 1, ZA, ZA, ZA, ZA, ZA, ZA, 0);
    v_transpose<<<elemBlocks_vt, 256, 0, stream>>>(qkvb, Vt);
    // S[b,h] = Q[b,h] @ K[b,h]^T   (batched over z = b*12+h)
    launch_gemm(stream, qkvb, qkvb + DD, nullptr, nullptr, Sb,
                NT, NT, HDIM, 3 * DD, 3 * DD, NT, BB * HH, HH,
                (long)NT * 3 * DD, (long)HDIM, (long)NT * 3 * DD, (long)HDIM,
                (long)HH * NT * NT, (long)NT * NT, 0);
    softmax_kernel<<<BB * HH * NT, 256, 0, stream>>>(Sb);
    // O[b, n, h*64+d] = P[b,h] @ V[b,h]   (W operand = Vt, [64, NT])
    launch_gemm(stream, Sb, Vt, nullptr, nullptr, Ob,
                NT, HDIM, NT, NT, NT, DD, BB * HH, HH,
                (long)HH * NT * NT, (long)NT * NT,
                (long)HH * HDIM * NT, (long)HDIM * NT,
                (long)NT * DD, (long)HDIM, 0);
    // tok += O @ proj_w^T + proj_b
    launch_gemm(stream, Ob, proj_w + (size_t)d * DD * DD,
                proj_b + (size_t)d * DD, tok, tok,
                ROWS, DD, DD, DD, DD, DD, 1, 1, ZA, ZA, ZA, ZA, ZA, ZA, 0);
    // ---- MLP ----
    layernorm_kernel<<<ROWS, 256, 0, stream>>>(tok, ln2_s + d * DD,
                                               ln2_b + d * DD, lnb);
    launch_gemm(stream, lnb, fc1_w + (size_t)d * FFD * DD,
                fc1_b + (size_t)d * FFD, nullptr, hb,
                ROWS, FFD, DD, DD, DD, FFD, 1, 1, ZA, ZA, ZA, ZA, ZA, ZA, 1);
    launch_gemm(stream, hb, fc2_w + (size_t)d * DD * FFD,
                fc2_b + (size_t)d * DD, tok, tok,
                ROWS, DD, FFD, FFD, FFD, DD, 1, 1, ZA, ZA, ZA, ZA, ZA, ZA, 0);
  }

  // head: out[b,:] = tok[b,0,:] @ head_w^T + head_b   (A row stride = NT*DD)
  launch_gemm(stream, tok, head_w, head_b, nullptr, out,
              BB, NCLS, DD, NT * DD, DD, NCLS, 1, 1, ZA, ZA, ZA, ZA, ZA, ZA, 0);
}
